// Encoder_49065706389648
// MI455X (gfx1250) — matrix operand order repaired and checked by reading, not compiled, not run on hardware
//
#include <hip/hip_runtime.h>
#include <hip/hip_bf16.h>

// ---------------------------------------------------------------------------
// Transformer encoder layer for MI455X (gfx1250, wave32, WMMA).
// All GEMMs run through v_wmma_f32_16x16x32_bf16 (bf16 in, f32 accum).
// Softmax / LayerNorm / bias / GELU epilogues in fp32.
// K-loop is software-pipelined (2x unroll, ping-pong fragment registers).
// ---------------------------------------------------------------------------

typedef __bf16 bf16_t;
typedef __attribute__((ext_vector_type(8)))  __bf16        v8bf;
typedef __attribute__((ext_vector_type(16))) __bf16        v16bf;
typedef __attribute__((ext_vector_type(8)))  float         v8f;
typedef __attribute__((ext_vector_type(8)))  unsigned short v8u16;

#define DIM   768
#define HID   1536
#define SEQ   2048
#define BATCH 8
#define ROWS  (BATCH * SEQ)   // 16384

__device__ __forceinline__ unsigned short f32_to_bf16_bits(float f) {
  unsigned int u = __float_as_uint(f);
  u += 0x7FFFu + ((u >> 16) & 1u);     // round-to-nearest-even
  return (unsigned short)(u >> 16);
}

// ---------------------------------------------------------------------------
// fp32 -> bf16 conversion (grid-stride)
// ---------------------------------------------------------------------------
__global__ __launch_bounds__(256)
void cvt_f32_bf16(const float* __restrict__ src, unsigned short* __restrict__ dst, int n) {
  int i = blockIdx.x * blockDim.x + threadIdx.x;
  int stride = gridDim.x * blockDim.x;
  for (; i < n; i += stride) dst[i] = f32_to_bf16_bits(src[i]);
}

// ---------------------------------------------------------------------------
// Generic GEMM: C[m,n] = scale * sum_k A[m,k] * W[n,k]  (+ bias[n])
// A: bf16 [M,K] row-major, W: bf16 [N,K] row-major (i.e. computes A @ W^T).
// Block: 256 threads = 8 waves (2 x 4). Block tile 128x128, wave tile 64x32.
// K must be a multiple of 64 (holds for 768 / 1536 / 2048).
// Epilogues: fp32 store / bf16 store / transposed-bf16 store / GELU+bf16.
// grid.z batches via element strides sA/sW/sC (stride 0 = shared operand).
// ---------------------------------------------------------------------------
enum { EPI_F32 = 0, EPI_BF16 = 1, EPI_BF16_T = 2, EPI_GELU_BF16 = 3 };

template <int EPI>
__global__ __launch_bounds__(256)
void gemm_bf16_wmma(const bf16_t* __restrict__ A, size_t sA,
                    const bf16_t* __restrict__ W, size_t sW,
                    const float* __restrict__ bias,
                    void* __restrict__ Cv, size_t sC,
                    int K, int lda, int ldw, int ldc, float scale)
{
  const int z = blockIdx.z;
  A += (size_t)z * sA;
  W += (size_t)z * sW;

  const int lane = threadIdx.x & 31;
  const int wave = threadIdx.x >> 5;
  const int wm = wave >> 2;          // 0..1
  const int wn = wave & 3;           // 0..3
  const int l15 = lane & 15;
  const int lh  = lane >> 4;         // half-wave select
  const int m0 = blockIdx.y * 128 + wm * 64;
  const int n0 = blockIdx.x * 128 + wn * 32;

  v8f acc[4][2] = {};

  // A fragment: lane l holds row m = l&15; K slots (lh*8 .. +7) and (lh*8+16 .. +23)
  const bf16_t* aptr[4];
#pragma unroll
  for (int mi = 0; mi < 4; ++mi)
    aptr[mi] = A + (size_t)(m0 + mi * 16 + l15) * lda + lh * 8;

  // B fragment: lane l holds col n = l&15; K slots (lh*8 .. +7) and (lh*8+16 .. +23)
  const bf16_t* bptr[2];
#pragma unroll
  for (int ni = 0; ni < 2; ++ni)
    bptr[ni] = W + (size_t)(n0 + ni * 16 + l15) * ldw + lh * 8;

  auto load_frags = [&](v16bf (&af)[4], v16bf (&bfm)[2], int kk) {
#pragma unroll
    for (int mi = 0; mi < 4; ++mi) {
      v8bf lo = *(const v8bf*)(aptr[mi] + kk);
      v8bf hi = *(const v8bf*)(aptr[mi] + kk + 16);
      af[mi] = __builtin_shufflevector(lo, hi, 0,1,2,3,4,5,6,7,8,9,10,11,12,13,14,15);
    }
#pragma unroll
    for (int ni = 0; ni < 2; ++ni) {
      v8bf lo = *(const v8bf*)(bptr[ni] + kk);
      v8bf hi = *(const v8bf*)(bptr[ni] + kk + 16);
      bfm[ni] = __builtin_shufflevector(lo, hi, 0,1,2,3,4,5,6,7,8,9,10,11,12,13,14,15);
    }
  };

  auto mma = [&](v16bf (&af)[4], v16bf (&bfm)[2]) {
#pragma unroll
    for (int mi = 0; mi < 4; ++mi)
#pragma unroll
      for (int ni = 0; ni < 2; ++ni)
        acc[mi][ni] = __builtin_amdgcn_wmma_f32_16x16x32_bf16(
            false, af[mi], false, bfm[ni], (short)0, acc[mi][ni], false, false);
  };

  // Software pipeline: prefetch the next 32-wide K slice while the current
  // slice's 8 WMMAs execute (ping-pong register fragments, K % 64 == 0).
  v16bf afA[4], bfA[2], afB[4], bfB[2];
  load_frags(afA, bfA, 0);
  for (int kk = 0; kk < K; kk += 64) {
    load_frags(afB, bfB, kk + 32);
    mma(afA, bfA);
    if (kk + 64 < K) load_frags(afA, bfA, kk + 64);
    mma(afB, bfB);
  }

  // Epilogue. C/D layout: lane l holds col n = l&15; vgpr r holds row (lh*8 + r).
#pragma unroll
  for (int mi = 0; mi < 4; ++mi) {
#pragma unroll
    for (int ni = 0; ni < 2; ++ni) {
      const int ng = n0 + ni * 16 + l15;
      const float bv = bias ? bias[ng] : 0.0f;
      const int mbase = m0 + mi * 16 + lh * 8;
      if constexpr (EPI == EPI_BF16_T) {
        // store C^T: element (m,n) -> Ct[n*ldc + m]; r-consecutive => contiguous
        v8u16 pack;
#pragma unroll
        for (int r = 0; r < 8; ++r)
          pack[r] = f32_to_bf16_bits(acc[mi][ni][r] * scale + bv);
        *(v8u16*)((unsigned short*)Cv + (size_t)z * sC + (size_t)ng * ldc + mbase) = pack;
      } else {
#pragma unroll
        for (int r = 0; r < 8; ++r) {
          float v = acc[mi][ni][r] * scale + bv;
          const size_t idx = (size_t)z * sC + (size_t)(mbase + r) * ldc + ng;
          if constexpr (EPI == EPI_F32) {
            ((float*)Cv)[idx] = v;
          } else if constexpr (EPI == EPI_BF16) {
            ((unsigned short*)Cv)[idx] = f32_to_bf16_bits(v);
          } else { // EPI_GELU_BF16, exact gelu
            v = 0.5f * v * (1.0f + erff(v * 0.70710678118654752f));
            ((unsigned short*)Cv)[idx] = f32_to_bf16_bits(v);
          }
        }
      }
    }
  }
}

// ---------------------------------------------------------------------------
// Row softmax over SEQ=2048 fp32 scores -> bf16 probabilities. 1 block / row.
// ---------------------------------------------------------------------------
__global__ __launch_bounds__(256)
void softmax_rows(const float* __restrict__ S, unsigned short* __restrict__ P) {
  __shared__ float red[8];
  const int row = blockIdx.x;
  const int tid = threadIdx.x;
  const float* s = S + (size_t)row * SEQ;

  float v[8];
  float mx = -1e30f;
#pragma unroll
  for (int j = 0; j < 8; ++j) { v[j] = s[tid + j * 256]; mx = fmaxf(mx, v[j]); }
#pragma unroll
  for (int off = 16; off > 0; off >>= 1) mx = fmaxf(mx, __shfl_xor(mx, off, 32));
  if ((tid & 31) == 0) red[tid >> 5] = mx;
  __syncthreads();
  mx = red[0];
#pragma unroll
  for (int i = 1; i < 8; ++i) mx = fmaxf(mx, red[i]);
  __syncthreads();

  float sum = 0.0f;
#pragma unroll
  for (int j = 0; j < 8; ++j) { v[j] = __expf(v[j] - mx); sum += v[j]; }
#pragma unroll
  for (int off = 16; off > 0; off >>= 1) sum += __shfl_xor(sum, off, 32);
  if ((tid & 31) == 0) red[tid >> 5] = sum;
  __syncthreads();
  sum = 0.0f;
#pragma unroll
  for (int i = 0; i < 8; ++i) sum += red[i];
  const float inv = 1.0f / sum;

  unsigned short* p = P + (size_t)row * SEQ;
#pragma unroll
  for (int j = 0; j < 8; ++j) p[tid + j * 256] = f32_to_bf16_bits(v[j] * inv);
}

// ---------------------------------------------------------------------------
// out = LayerNorm(X + Y) * gamma + beta  over DIM=768. 1 block / row.
// Optionally also writes a bf16 copy (for the next GEMM's A operand).
// ---------------------------------------------------------------------------
__global__ __launch_bounds__(256)
void add_layernorm(const float* __restrict__ X, const float* __restrict__ Y,
                   const float* __restrict__ gamma, const float* __restrict__ beta,
                   float* __restrict__ outf, unsigned short* __restrict__ outb) {
  __shared__ float reds[8], redq[8];
  const int row = blockIdx.x;
  const int tid = threadIdx.x;
  const float* x = X + (size_t)row * DIM;
  const float* y = Y + (size_t)row * DIM;

  float v[3];
  float s = 0.0f, q = 0.0f;
#pragma unroll
  for (int j = 0; j < 3; ++j) {
    v[j] = x[tid + j * 256] + y[tid + j * 256];
    s += v[j];
    q += v[j] * v[j];
  }
#pragma unroll
  for (int off = 16; off > 0; off >>= 1) {
    s += __shfl_xor(s, off, 32);
    q += __shfl_xor(q, off, 32);
  }
  if ((tid & 31) == 0) { reds[tid >> 5] = s; redq[tid >> 5] = q; }
  __syncthreads();
  s = 0.0f; q = 0.0f;
#pragma unroll
  for (int i = 0; i < 8; ++i) { s += reds[i]; q += redq[i]; }

  const float mu   = s * (1.0f / DIM);
  const float var  = q * (1.0f / DIM) - mu * mu;
  const float rstd = rsqrtf(var + 1e-12f);

#pragma unroll
  for (int j = 0; j < 3; ++j) {
    const int c = tid + j * 256;
    const float o = (v[j] - mu) * rstd * gamma[c] + beta[c];
    outf[(size_t)row * DIM + c] = o;
    if (outb) outb[(size_t)row * DIM + c] = f32_to_bf16_bits(o);
  }
}

// ---------------------------------------------------------------------------
// Host-side orchestration
// ---------------------------------------------------------------------------
extern "C" void kernel_launch(void* const* d_in, const int* in_sizes, int n_in,
                              void* d_out, int out_size, void* d_ws, size_t ws_size,
                              hipStream_t stream) {
  (void)in_sizes; (void)n_in; (void)out_size; (void)ws_size;

  const float* x   = (const float*)d_in[0];
  const float* Wq  = (const float*)d_in[1];
  const float* bq  = (const float*)d_in[2];
  const float* Wk  = (const float*)d_in[3];
  const float* bk  = (const float*)d_in[4];
  const float* Wv  = (const float*)d_in[5];
  const float* bv  = (const float*)d_in[6];
  const float* Wp  = (const float*)d_in[7];
  const float* bp  = (const float*)d_in[8];
  const float* g1  = (const float*)d_in[9];
  const float* be1 = (const float*)d_in[10];
  const float* W1  = (const float*)d_in[11];
  const float* b1  = (const float*)d_in[12];
  const float* W2  = (const float*)d_in[13];
  const float* b2  = (const float*)d_in[14];
  const float* g2  = (const float*)d_in[15];
  const float* be2 = (const float*)d_in[16];

  // ---- workspace layout (all chunk sizes are multiples of 256B) -----------
  char* ws = (char*)d_ws;
  size_t off = 0;
  auto alloc = [&](size_t bytes) { size_t o = off; off += (bytes + 255) & ~(size_t)255; return o; };

  const size_t o_wqb = alloc((size_t)DIM * DIM * 2);
  const size_t o_wkb = alloc((size_t)DIM * DIM * 2);
  const size_t o_wvb = alloc((size_t)DIM * DIM * 2);
  const size_t o_wpb = alloc((size_t)DIM * DIM * 2);
  const size_t o_w1b = alloc((size_t)HID * DIM * 2);
  const size_t o_w2b = alloc((size_t)DIM * HID * 2);
  const size_t o_xb  = alloc((size_t)ROWS * DIM * 2);     // 25.2 MB (dead after V proj)
  const size_t o_qb  = alloc((size_t)ROWS * DIM * 2);     // dead after scores
  const size_t o_kb  = alloc((size_t)ROWS * DIM * 2);     // dead after scores
  const size_t o_vt  = alloc((size_t)ROWS * DIM * 2);     // V^T per batch [768,2048]
  const size_t o_sc  = alloc((size_t)SEQ * SEQ * 4);      // per-batch scores (reused)
  const size_t o_p   = alloc((size_t)SEQ * SEQ * 2);      // per-batch probs  (reused)
  const size_t o_at  = alloc((size_t)ROWS * DIM * 2);     // attn (bf16)
  const size_t o_hf  = alloc((size_t)ROWS * DIM * 4);     // h (fp32, residual)
  // Aliases over dead regions (stream order guarantees safety):
  const size_t o_proj = o_xb;   // fp32 50.3 MB over xb+qb   (proj out, then mlp2 out)
  const size_t o_hb   = o_kb;   // bf16 25.2 MB over kb      (h in bf16)
  const size_t o_mid  = o_vt;   // bf16 50.3 MB over vt+sc+p (gelu hidden)
  // total footprint ~211 MB

  unsigned short* xb   = (unsigned short*)(ws + o_xb);
  unsigned short* wqb  = (unsigned short*)(ws + o_wqb);
  unsigned short* wkb  = (unsigned short*)(ws + o_wkb);
  unsigned short* wvb  = (unsigned short*)(ws + o_wvb);
  unsigned short* wpb  = (unsigned short*)(ws + o_wpb);
  unsigned short* w1b  = (unsigned short*)(ws + o_w1b);
  unsigned short* w2b  = (unsigned short*)(ws + o_w2b);
  unsigned short* qb   = (unsigned short*)(ws + o_qb);
  unsigned short* kb   = (unsigned short*)(ws + o_kb);
  unsigned short* vt   = (unsigned short*)(ws + o_vt);
  float*          sc   = (float*)         (ws + o_sc);
  unsigned short* pb   = (unsigned short*)(ws + o_p);
  unsigned short* at   = (unsigned short*)(ws + o_at);
  float*          hf   = (float*)         (ws + o_hf);
  float*          proj = (float*)         (ws + o_proj);
  unsigned short* hb   = (unsigned short*)(ws + o_hb);
  unsigned short* mid  = (unsigned short*)(ws + o_mid);
  float*          m2f  = (float*)         (ws + o_proj);  // mlp2 fp32, reuse proj

  const dim3 blk(256);

  // ---- 1) fp32 -> bf16 conversions ---------------------------------------
  cvt_f32_bf16<<<dim3(2048), blk, 0, stream>>>(x,  xb,  ROWS * DIM);
  cvt_f32_bf16<<<dim3(128),  blk, 0, stream>>>(Wq, wqb, DIM * DIM);
  cvt_f32_bf16<<<dim3(128),  blk, 0, stream>>>(Wk, wkb, DIM * DIM);
  cvt_f32_bf16<<<dim3(128),  blk, 0, stream>>>(Wv, wvb, DIM * DIM);
  cvt_f32_bf16<<<dim3(128),  blk, 0, stream>>>(Wp, wpb, DIM * DIM);
  cvt_f32_bf16<<<dim3(256),  blk, 0, stream>>>(W1, w1b, HID * DIM);
  cvt_f32_bf16<<<dim3(256),  blk, 0, stream>>>(W2, w2b, DIM * HID);

  // ---- 2) Q, K projections; V projection stored transposed per batch -----
  gemm_bf16_wmma<EPI_BF16><<<dim3(6, 128, 1), blk, 0, stream>>>(
      (const bf16_t*)xb, 0, (const bf16_t*)wqb, 0, bq, qb, 0,
      DIM, DIM, DIM, DIM, 1.0f);
  gemm_bf16_wmma<EPI_BF16><<<dim3(6, 128, 1), blk, 0, stream>>>(
      (const bf16_t*)xb, 0, (const bf16_t*)wkb, 0, bk, kb, 0,
      DIM, DIM, DIM, DIM, 1.0f);
  gemm_bf16_wmma<EPI_BF16_T><<<dim3(6, 16, BATCH), blk, 0, stream>>>(
      (const bf16_t*)xb, (size_t)SEQ * DIM, (const bf16_t*)wvb, 0, bv,
      vt, (size_t)DIM * SEQ, DIM, DIM, DIM, SEQ /*ldc of V^T*/, 1.0f);

  // ---- 3) attention per batch: scores -> softmax -> P @ V^T --------------
  const float inv_sqrt_d = 0.03608439182435161f; // 1/sqrt(768)
  for (int b = 0; b < BATCH; ++b) {
    const bf16_t* qbB = (const bf16_t*)(qb + (size_t)b * SEQ * DIM);
    const bf16_t* kbB = (const bf16_t*)(kb + (size_t)b * SEQ * DIM);
    const bf16_t* vtB = (const bf16_t*)(vt + (size_t)b * DIM * SEQ);
    unsigned short* atB = at + (size_t)b * SEQ * DIM;

    // scores[q,k] = (Q . K) / sqrt(d)   (W = K, same A@W^T pattern)
    gemm_bf16_wmma<EPI_F32><<<dim3(16, 16, 1), blk, 0, stream>>>(
        qbB, 0, kbB, 0, nullptr, sc, 0, DIM, DIM, DIM, SEQ, inv_sqrt_d);
    softmax_rows<<<dim3(SEQ), blk, 0, stream>>>(sc, pb);
    // attn[q,d] = sum_k P[q,k] * Vt[d,k]
    gemm_bf16_wmma<EPI_BF16><<<dim3(6, 16, 1), blk, 0, stream>>>(
        (const bf16_t*)pb, 0, vtB, 0, nullptr, atB, 0, SEQ, SEQ, SEQ, DIM, 1.0f);
  }

  // ---- 4) output projection + residual + LN1 -----------------------------
  gemm_bf16_wmma<EPI_F32><<<dim3(6, 128, 1), blk, 0, stream>>>(
      (const bf16_t*)at, 0, (const bf16_t*)wpb, 0, bp, proj, 0,
      DIM, DIM, DIM, DIM, 1.0f);
  add_layernorm<<<dim3(ROWS), blk, 0, stream>>>(x, proj, g1, be1, hf, hb);

  // ---- 5) MLP: gelu(h @ W1^T + b1) @ W2^T + b2, residual + LN2 -----------
  gemm_bf16_wmma<EPI_GELU_BF16><<<dim3(12, 128, 1), blk, 0, stream>>>(
      (const bf16_t*)hb, 0, (const bf16_t*)w1b, 0, b1, mid, 0,
      DIM, DIM, DIM, HID, 1.0f);
  gemm_bf16_wmma<EPI_F32><<<dim3(6, 128, 1), blk, 0, stream>>>(
      (const bf16_t*)mid, 0, (const bf16_t*)w2b, 0, b2, m2f, 0,
      HID, HID, HID, DIM, 1.0f);
  add_layernorm<<<dim3(ROWS), blk, 0, stream>>>(hf, m2f, g2, be2,
                                                (float*)d_out, nullptr);
}
